// MySwin_no_projection_50929722196706
// MI455X (gfx1250) — matrix operand order repaired and checked by reading, compiled, not run on hardware
//
#include <hip/hip_runtime.h>
#include <math.h>

// ---------------------------------------------------------------------------
// Swin-T forward for MI455X (gfx1250, wave32, WMMA).
// All dense GEMMs run through v_wmma_f32_16x16x32_bf16 (bf16 in, f32 acc).
// Weight (B) panels are staged into LDS once per workgroup via the gfx1250
// async-copy path (global_load_async_to_lds_b128 + s_wait_asynccnt), double
// buffered. Epilogues (bias / GELU / residual-add) fused into the GEMM.
// Window attention (WIN=2 -> 4-token windows) is a small VALU kernel.
// ---------------------------------------------------------------------------

typedef __attribute__((ext_vector_type(16))) __bf16 v16bf;
typedef __attribute__((ext_vector_type(8)))  __bf16 v8bf;
typedef __attribute__((ext_vector_type(8)))  float  v8f;
typedef int v4i __attribute__((vector_size(16)));

enum {
  EPI_BIAS_BF16       = 0,  // bOut = acc + bias            (qkv, head1)
  EPI_BIAS_GELU_BF16  = 1,  // bOut = gelu(acc + bias)      (fc1)
  EPI_BIAS_RESADD_F32 = 2,  // fOut += acc + bias           (proj, fc2)
  EPI_NOBIAS_F32      = 3,  // fOut  = acc                  (patch-merge reduction)
  EPI_BIAS_F32        = 4   // fOut  = acc + bias           (head2 -> d_out)
};

#if __has_builtin(__builtin_amdgcn_global_load_async_to_lds_b128)
#define HAS_ASYNC_LDS 1
#else
#define HAS_ASYNC_LDS 0
#endif

// Stage one 16-byte chunk global -> LDS.
// Builtin prototype (from clang): (v4i addrspace(1)*, v4i addrspace(3)*, imm, imm)
__device__ __forceinline__ void stage_b128(const __bf16* g, __bf16* l) {
#if HAS_ASYNC_LDS
  __builtin_amdgcn_global_load_async_to_lds_b128(
      (__attribute__((address_space(1))) v4i*)g,
      (__attribute__((address_space(3))) v4i*)l, 0, 0);
#else
  *(v8bf*)l = *(const v8bf*)g;    // sync fallback: global_load + ds_store
#endif
}

__device__ __forceinline__ void wait_async_copies() {
#if HAS_ASYNC_LDS
#if __has_builtin(__builtin_amdgcn_s_wait_asynccnt)
  __builtin_amdgcn_s_wait_asynccnt(0);
#else
  asm volatile("s_wait_asynccnt 0" ::: "memory");
#endif
#endif
}

__device__ __forceinline__ float gelu_exact(float x) {
  return 0.5f * x * (1.0f + erff(x * 0.70710678118654752f));
}

// ---------------------------------------------------------------------------
// GEMM: out[M,N] = A[M,K](bf16) x W[N,K](bf16)^T  (+ epilogue)
// One wave -> 32x32 tile (2x2 WMMA accumulators). 4 waves / block along M;
// the 32-col weight panel is shared through LDS (async copy, double buffer).
// Requires M % 32 == 0, N % 32 == 0, K % 32 == 0 (true for every GEMM here).
// ---------------------------------------------------------------------------
template<int EPI>
__global__ __launch_bounds__(128)
void gemm_bf16_wmma(const __bf16* __restrict__ A, const __bf16* __restrict__ W,
                    const float* __restrict__ bias,
                    float* __restrict__ fOut, __bf16* __restrict__ bOut,
                    int M, int N, int K)
{
  __shared__ __bf16 Bpanel[2][32 * 32];    // double-buffered 32(N) x 32(K) panel

  const int tid  = threadIdx.x;
  const int lane = tid & 31;
  const int wave = tid >> 5;
  const int m0 = (blockIdx.y * 4 + wave) * 32;
  const int n0 = blockIdx.x * 32;
  const bool active = (m0 < M);            // inactive waves still stage/barrier

  const int lr = lane & 15;
  const int g  = lane >> 4;
  const int ka = g << 3;                   // A lane-group chunk base (ISA 16-bit A 16x32)
  const int kb = g << 4;                   // unused; B frag below uses ka, same k order as A

  const int mA = active ? m0 : 0;          // clamp A rows for inactive waves
  const __bf16* a0p = A + (size_t)(mA + lr) * K;
  const __bf16* a1p = a0p + (size_t)16 * K;

  // Staging map: thread t copies 8 bf16: row = t>>2 (0..31), seg = t&3 (0..3)
  const int srow = tid >> 2;
  const int scol = (tid & 3) << 3;
  const __bf16* wsrc = W + (size_t)(n0 + srow) * K + scol;

  v8f acc00 = {}, acc01 = {}, acc10 = {}, acc11 = {};

  int buf = 0;
  stage_b128(wsrc, &Bpanel[0][srow * 32 + scol]);

  for (int k = 0; k < K; k += 32) {
    wait_async_copies();                   // this wave's copies into Bpanel[buf] landed
    __syncthreads();                       // -> whole panel visible to all waves

    if (k + 32 < K) {                      // prefetch next panel into the other buffer
      stage_b128(wsrc + k + 32, &Bpanel[buf ^ 1][srow * 32 + scol]);
      __builtin_prefetch(a0p + k + 32, 0, 1);
      __builtin_prefetch(a1p + k + 32, 0, 1);
    }

    union { v16bf v; v8bf h[2]; } a0, a1, b0, b1;
    // A 16x32 bf16 frag from global: lane elems = K[ka..ka+7], K[16+ka..+7]
    a0.h[0] = *(const v8bf*)(a0p + k + ka);
    a0.h[1] = *(const v8bf*)(a0p + k + ka + 16);
    a1.h[0] = *(const v8bf*)(a1p + k + ka);
    a1.h[1] = *(const v8bf*)(a1p + k + ka + 16);
    // B 32x16 bf16 frag from LDS: lane elems = K[ka..ka+7], K[16+ka..+7]
    {
      const __bf16* bp0 = &Bpanel[buf][lr * 32 + ka];
      const __bf16* bp1 = bp0 + 16 * 32;
      b0.h[0] = *(const v8bf*)(bp0);
      b0.h[1] = *(const v8bf*)(bp0 + 16);
      b1.h[0] = *(const v8bf*)(bp1);
      b1.h[1] = *(const v8bf*)(bp1 + 16);
    }

    acc00 = __builtin_amdgcn_wmma_f32_16x16x32_bf16(false, a0.v, false, b0.v, (short)0, acc00, false, false);
    acc01 = __builtin_amdgcn_wmma_f32_16x16x32_bf16(false, a0.v, false, b1.v, (short)0, acc01, false, false);
    acc10 = __builtin_amdgcn_wmma_f32_16x16x32_bf16(false, a1.v, false, b0.v, (short)0, acc10, false, false);
    acc11 = __builtin_amdgcn_wmma_f32_16x16x32_bf16(false, a1.v, false, b1.v, (short)0, acc11, false, false);

    buf ^= 1;
  }

  if (!active) return;                     // after all barriers: safe to exit

  // C/D layout: element r of v8f -> row = r + 8*g, col = lane&15
  const int col0  = n0 + lr;
  const int rbase = m0 + (g << 3);
  float bias0 = 0.f, bias1 = 0.f;
  if (EPI != EPI_NOBIAS_F32) { bias0 = bias[col0]; bias1 = bias[col0 + 16]; }

#pragma unroll
  for (int r = 0; r < 8; ++r) {
    const int row0 = rbase + r;
    const int row1 = row0 + 16;
    float v00 = acc00[r], v01 = acc01[r], v10 = acc10[r], v11 = acc11[r];
    const size_t i00 = (size_t)row0 * N + col0;
    const size_t i01 = i00 + 16;
    const size_t i10 = (size_t)row1 * N + col0;
    const size_t i11 = i10 + 16;
    if (EPI == EPI_BIAS_BF16) {
      bOut[i00] = (__bf16)(v00 + bias0); bOut[i01] = (__bf16)(v01 + bias1);
      bOut[i10] = (__bf16)(v10 + bias0); bOut[i11] = (__bf16)(v11 + bias1);
    } else if (EPI == EPI_BIAS_GELU_BF16) {
      bOut[i00] = (__bf16)gelu_exact(v00 + bias0); bOut[i01] = (__bf16)gelu_exact(v01 + bias1);
      bOut[i10] = (__bf16)gelu_exact(v10 + bias0); bOut[i11] = (__bf16)gelu_exact(v11 + bias1);
    } else if (EPI == EPI_BIAS_RESADD_F32) {
      fOut[i00] += v00 + bias0; fOut[i01] += v01 + bias1;
      fOut[i10] += v10 + bias0; fOut[i11] += v11 + bias1;
    } else if (EPI == EPI_NOBIAS_F32) {
      fOut[i00] = v00; fOut[i01] = v01; fOut[i10] = v10; fOut[i11] = v11;
    } else { // EPI_BIAS_F32
      fOut[i00] = v00 + bias0; fOut[i01] = v01 + bias1;
      fOut[i10] = v10 + bias0; fOut[i11] = v11 + bias1;
    }
  }
}

// ---------------------------------------------------------------------------
// Patch embed: 2x2 stride-2 conv (K=12) over NCHW input. out[t, c], fp32.
// ---------------------------------------------------------------------------
__global__ __launch_bounds__(256)
void patch_conv_kernel(const float* __restrict__ x, const float* __restrict__ w,
                       const float* __restrict__ bias, float* __restrict__ out)
{
  const int T = 64 * 48 * 48;
  int idx = blockIdx.x * blockDim.x + threadIdx.x;
  if (idx >= T * 96) return;
  int c = idx % 96;
  int t = idx / 96;
  int xc = t % 48;
  int yr = (t / 48) % 48;
  int b  = t / (48 * 48);
  float acc = bias[c];
  const float* wp = w + c * 12;
#pragma unroll
  for (int ci = 0; ci < 3; ++ci)
#pragma unroll
    for (int ph = 0; ph < 2; ++ph)
#pragma unroll
      for (int pw = 0; pw < 2; ++pw)
        acc += x[(((size_t)b * 3 + ci) * 96 + (2 * yr + ph)) * 96 + (2 * xc + pw)]
               * wp[ci * 4 + ph * 2 + pw];
  out[idx] = acc;
}

// ---------------------------------------------------------------------------
// LayerNorm: one wave32 per token, shfl_xor reduction. OutT = float or __bf16.
// ---------------------------------------------------------------------------
template<typename OutT>
__global__ __launch_bounds__(256)
void layernorm_kernel(const float* __restrict__ in, const float* __restrict__ g,
                      const float* __restrict__ b, OutT* __restrict__ out,
                      int M, int C)
{
  int token = blockIdx.x * 8 + (threadIdx.x >> 5);
  int lane  = threadIdx.x & 31;
  if (token >= M) return;
  const float* row = in + (size_t)token * C;
  float s = 0.f;
  for (int c = lane; c < C; c += 32) s += row[c];
#pragma unroll
  for (int m = 16; m; m >>= 1) s += __shfl_xor(s, m, 32);
  float mean = s / (float)C;
  float v = 0.f;
  for (int c = lane; c < C; c += 32) { float d = row[c] - mean; v += d * d; }
#pragma unroll
  for (int m = 16; m; m >>= 1) v += __shfl_xor(v, m, 32);
  float inv = rsqrtf(v / (float)C + 1e-5f);
  OutT* orow = out + (size_t)token * C;
  for (int c = lane; c < C; c += 32)
    orow[c] = (OutT)((row[c] - mean) * inv * g[c] + b[c]);
}

// ---------------------------------------------------------------------------
// Window attention, WIN=2 (N=4 tokens), head_dim=32 everywhere.
// One thread = one (window, head, query-token). Shift folded into indexing:
// rolled (y,x) reads token ((y+shift)%H, (x+shift)%W); output written back to
// the same natural token index, which is exactly roll-then-unroll.
// qkv row layout: [q(heads*32) | k(heads*32) | v(heads*32)].
// ---------------------------------------------------------------------------
__global__ __launch_bounds__(256)
void win_attn_kernel(const __bf16* __restrict__ qkv, const float* __restrict__ relb,
                     __bf16* __restrict__ out, int B, int H, int W,
                     int heads, int shift, int C)
{
  const int nh = H >> 1, nw = W >> 1;
  const int total = B * nh * nw * heads * 4;
  int tid = blockIdx.x * blockDim.x + threadIdx.x;
  if (tid >= total) return;
  const int i   = tid & 3;
  const int h   = (tid >> 2) % heads;
  const int win = tid / (heads << 2);
  const int ww  = win % nw;
  const int wh  = (win / nw) % nh;
  const int b   = win / (nw * nh);
  const int s3C = 3 * C;

  int tok[4], lab[4];
#pragma unroll
  for (int j = 0; j < 4; ++j) {
    int yw = wh * 2 + (j >> 1);
    int xw = ww * 2 + (j & 1);
    int y = yw + shift; if (y >= H) y -= H;
    int x = xw + shift; if (x >= W) x -= W;
    tok[j] = (b * H + y) * W + x;
    int ly = (yw < H - 2) ? 0 : ((yw < H - 1) ? 1 : 2);
    int lx = (xw < W - 2) ? 0 : ((xw < W - 1) ? 1 : 2);
    lab[j] = ly * 3 + lx;
  }

  float q[32];
  {
    const __bf16* qp = qkv + (size_t)tok[i] * s3C + h * 32;
#pragma unroll
    for (int d = 0; d < 32; ++d) q[d] = (float)qp[d];
  }
  const float scale = 0.17677669529663687f;  // 32^-0.5
  float sc[4];
#pragma unroll
  for (int j = 0; j < 4; ++j) {
    const __bf16* kp = qkv + (size_t)tok[j] * s3C + C + h * 32;
    float a = 0.f;
#pragma unroll
    for (int d = 0; d < 32; ++d) a += q[d] * (float)kp[d];
    a *= scale;
    int ri = 3 * ((i >> 1) - (j >> 1) + 1) + ((i & 1) - (j & 1) + 1);
    a += relb[ri * heads + h];
    if (shift && lab[i] != lab[j]) a -= 100.0f;
    sc[j] = a;
  }
  float mx = fmaxf(fmaxf(sc[0], sc[1]), fmaxf(sc[2], sc[3]));
  float p[4], sum = 0.f;
#pragma unroll
  for (int j = 0; j < 4; ++j) { p[j] = expf(sc[j] - mx); sum += p[j]; }
  float rs = 1.f / sum;
  float o[32];
#pragma unroll
  for (int d = 0; d < 32; ++d) o[d] = 0.f;
#pragma unroll
  for (int j = 0; j < 4; ++j) {
    const __bf16* vp = qkv + (size_t)tok[j] * s3C + 2 * C + h * 32;
    float pj = p[j] * rs;
#pragma unroll
    for (int d = 0; d < 32; ++d) o[d] += pj * (float)vp[d];
  }
  __bf16* op = out + (size_t)tok[i] * C + h * 32;
#pragma unroll
  for (int d = 0; d < 32; ++d) op[d] = (__bf16)o[d];
}

// ---------------------------------------------------------------------------
// Patch-merge gather: out[b,y2,x2, sel*C + c] = in[b, 2y2+dy, 2x2+dx, c]
// with (dy,dx) = (0,0),(1,0),(0,1),(1,1) for sel = 0..3 (x0,x1,x2,x3 order).
// ---------------------------------------------------------------------------
__global__ __launch_bounds__(256)
void merge_gather_kernel(const float* __restrict__ in, float* __restrict__ out,
                         int B, int H, int W, int C)
{
  const int H2 = H >> 1, W2 = W >> 1, C4 = 4 * C;
  const size_t total = (size_t)B * H2 * W2 * C4;
  size_t idx = blockIdx.x * (size_t)blockDim.x + threadIdx.x;
  if (idx >= total) return;
  int c4 = (int)(idx % C4);
  size_t t = idx / C4;
  int x2 = (int)(t % W2);
  int y2 = (int)((t / W2) % H2);
  int b  = (int)(t / ((size_t)W2 * H2));
  int sel = c4 / C, c = c4 % C;
  int dy = sel & 1;
  int dx = (sel >> 1) & 1;
  out[idx] = in[(((size_t)b * H + 2 * y2 + dy) * W + (2 * x2 + dx)) * C + c];
}

__global__ __launch_bounds__(256)
void mean_pool_kernel(const __bf16* __restrict__ in, __bf16* __restrict__ out,
                      int B, int T, int C)
{
  int idx = blockIdx.x * blockDim.x + threadIdx.x;
  if (idx >= B * C) return;
  int c = idx % C, b = idx / C;
  float s = 0.f;
  for (int t = 0; t < T; ++t) s += (float)in[((size_t)b * T + t) * C + c];
  out[idx] = (__bf16)(s / (float)T);
}

__global__ __launch_bounds__(256)
void f32_to_bf16_kernel(const float* __restrict__ in, __bf16* __restrict__ out, int n)
{
  int i = blockIdx.x * blockDim.x + threadIdx.x;
  if (i < n) out[i] = (__bf16)in[i];
}

// ---------------------------------------------------------------------------
// Host orchestration.
// Param leaves assumed flattened depth-first in dict-insertion order:
//   0:x  1:patch_w 2:patch_b 3:patch_norm.g 4:patch_norm.b
//   blocks (12 of them, 13 leaves each, base = 5 + 13*blockLinear):
//     +0 n1.g +1 n1.b +2 qkv_w +3 qkv_b +4 rel_bias +5 proj_w +6 proj_b
//     +7 n2.g +8 n2.b +9 fc1_w +10 fc1_b +11 fc2_w +12 fc2_b
//   merges (base = 161 + 3*s): +0 norm.g +1 norm.b +2 red_w
//   170 norm.g 171 norm.b 172 head_w 173 head_b
// ---------------------------------------------------------------------------
static inline dim3 gemm_grid(int M, int N) { return dim3(N / 32, (M + 127) / 128); }

extern "C" void kernel_launch(void* const* d_in, const int* in_sizes, int n_in,
                              void* d_out, int out_size, void* d_ws, size_t ws_size,
                              hipStream_t stream)
{
  (void)in_sizes; (void)n_in; (void)out_size; (void)ws_size;
  const int B = 64;
  static const int CS[4]     = {96, 192, 384, 768};
  static const int HSZ[4]    = {48, 24, 12, 6};
  static const int DEPTH[4]  = {2, 2, 6, 2};
  static const int NHEADS[4] = {3, 6, 12, 24};

  const float* xin = (const float*)d_in[0];
  auto P = [&](int i) { return (const float*)d_in[i]; };

  // Workspace carve (all sized for stage 0 worst cases; ~320 MB total).
  char* ws = (char*)d_ws;
  size_t off = 0;
  auto carve = [&](size_t bytes) -> void* {
    void* p = ws + off;
    off += (bytes + 255) & ~(size_t)255;
    return p;
  };
  const size_t M0 = (size_t)B * 48 * 48;                       // 147456 tokens
  float*  xres    = (float*) carve(M0 * 96  * sizeof(float));  // residual stream
  __bf16* lnbuf   = (__bf16*)carve(M0 * 96  * sizeof(__bf16)); // LN outputs
  __bf16* qkvbuf  = (__bf16*)carve(M0 * 288 * sizeof(__bf16));
  __bf16* attnout = (__bf16*)carve(M0 * 96  * sizeof(__bf16));
  __bf16* hidden  = (__bf16*)carve(M0 * 384 * sizeof(__bf16)); // MLP hidden
  __bf16* wscr    = (__bf16*)carve((size_t)3072 * 768 * sizeof(__bf16)); // weight slab
  __bf16* pooled  = (__bf16*)carve((size_t)B * 768 * sizeof(__bf16));
  __bf16* h1      = (__bf16*)carve((size_t)B * 768 * sizeof(__bf16));
  float*  tmpf    = (float*)hidden;  // overlay: conv/gather tmp never live with MLP hidden

  // Stream-ordered weight conversion into the shared slab (safe: all GEMMs
  // that read wscr are enqueued before the next conversion overwrites it).
  auto convW = [&](int leaf, int n) {
    f32_to_bf16_kernel<<<(n + 255) / 256, 256, 0, stream>>>(P(leaf), wscr, n);
  };

  // ---- patch embed + patch_norm -> xres (fp32) ----
  {
    int n = (int)(M0 * 96);
    patch_conv_kernel<<<(n + 255) / 256, 256, 0, stream>>>(xin, P(1), P(2), tmpf);
    layernorm_kernel<float><<<((int)M0 + 7) / 8, 256, 0, stream>>>(
        tmpf, P(3), P(4), xres, (int)M0, 96);
  }

  // ---- stages ----
  int blin = 0;
  for (int s = 0; s < 4; ++s) {
    const int C = CS[s], H = HSZ[s], W = HSZ[s];
    const int M = B * H * W;
    for (int d = 0; d < DEPTH[s]; ++d, ++blin) {
      const int base = 5 + 13 * blin;
      const int shift = (d & 1) ? 1 : 0;  // WIN/2 = 1

      // x = x + proj(attn(LN1(x)))
      layernorm_kernel<__bf16><<<(M + 7) / 8, 256, 0, stream>>>(
          xres, P(base + 0), P(base + 1), lnbuf, M, C);
      convW(base + 2, 3 * C * C);
      gemm_bf16_wmma<EPI_BIAS_BF16><<<gemm_grid(M, 3 * C), 128, 0, stream>>>(
          lnbuf, wscr, P(base + 3), nullptr, qkvbuf, M, 3 * C, C);
      {
        int at = M * NHEADS[s];
        win_attn_kernel<<<(at + 255) / 256, 256, 0, stream>>>(
            qkvbuf, P(base + 4), attnout, B, H, W, NHEADS[s], shift, C);
      }
      convW(base + 5, C * C);
      gemm_bf16_wmma<EPI_BIAS_RESADD_F32><<<gemm_grid(M, C), 128, 0, stream>>>(
          attnout, wscr, P(base + 6), xres, nullptr, M, C, C);

      // x = x + fc2(gelu(fc1(LN2(x))))
      layernorm_kernel<__bf16><<<(M + 7) / 8, 256, 0, stream>>>(
          xres, P(base + 7), P(base + 8), lnbuf, M, C);
      convW(base + 9, 4 * C * C);
      gemm_bf16_wmma<EPI_BIAS_GELU_BF16><<<gemm_grid(M, 4 * C), 128, 0, stream>>>(
          lnbuf, wscr, P(base + 10), nullptr, hidden, M, 4 * C, C);
      convW(base + 11, 4 * C * C);
      gemm_bf16_wmma<EPI_BIAS_RESADD_F32><<<gemm_grid(M, C), 128, 0, stream>>>(
          hidden, wscr, P(base + 12), xres, nullptr, M, C, 4 * C);
    }
    if (s < 3) {
      const int mbase = 161 + 3 * s;
      const int M4 = M / 4, C4 = 4 * C;
      size_t n = (size_t)M4 * C4;
      merge_gather_kernel<<<(unsigned)((n + 255) / 256), 256, 0, stream>>>(
          xres, tmpf, B, H, W, C);
      layernorm_kernel<__bf16><<<(M4 + 7) / 8, 256, 0, stream>>>(
          tmpf, P(mbase + 0), P(mbase + 1), lnbuf, M4, C4);
      convW(mbase + 2, 2 * C * C4);
      gemm_bf16_wmma<EPI_NOBIAS_F32><<<gemm_grid(M4, 2 * C), 128, 0, stream>>>(
          lnbuf, wscr, nullptr, xres, nullptr, M4, 2 * C, C4);
    }
  }

  // ---- final norm, global mean pool, double head GEMM ----
  {
    const int M = B * 6 * 6;  // 2304 tokens, C = 768
    layernorm_kernel<__bf16><<<(M + 7) / 8, 256, 0, stream>>>(
        xres, P(170), P(171), lnbuf, M, 768);
    int n = B * 768;
    mean_pool_kernel<<<(n + 255) / 256, 256, 0, stream>>>(lnbuf, pooled, B, 36, 768);
    convW(172, 768 * 768);
    gemm_bf16_wmma<EPI_BIAS_BF16><<<gemm_grid(B, 768), 128, 0, stream>>>(
        pooled, wscr, P(173), nullptr, h1, B, 768, 768);
    gemm_bf16_wmma<EPI_BIAS_F32><<<gemm_grid(B, 768), 128, 0, stream>>>(
        h1, wscr, P(173), (float*)d_out, nullptr, B, 768, 768);
  }
}
